// SpatioTemporalPerformerEncoder_89258010345705
// MI455X (gfx1250) — matrix operand order repaired and checked by reading, not compiled, not run on hardware
//
#include <hip/hip_runtime.h>
#include <hip/hip_bf16.h>

#define N_DEPTH 4
#define N_HEADS 8
#define N_DIM   512
#define N_DH    64
#define N_MF    256
#define N_FF    2048
#define N_SEQ   4096
#define N_BATCH 2
#define N_ROWS  (N_BATCH * N_SEQ)      // 8192
#define EPSK    1e-4f
#define LNEPS   1e-5f
#define RATIO   0.0625f                // 256^-0.5
#define DNRM    0.35355339059327373f   // 64^-0.25

#define USE_ASYNC 1                    // gfx1250 GLOBAL_LOAD_ASYNC_TO_LDS path

typedef __bf16 bf16;
typedef __attribute__((ext_vector_type(16))) __bf16 v16bf;
typedef __attribute__((ext_vector_type(8)))  __bf16 v8bf;
typedef __attribute__((ext_vector_type(8)))  float  v8f;

static __device__ __forceinline__ v16bf cat8(v8bf lo, v8bf hi) {
  return __builtin_shufflevector(lo, hi, 0,1,2,3,4,5,6,7,8,9,10,11,12,13,14,15);
}

// gfx1250 async 16B copy global -> LDS (ASYNCcnt-tracked, no VGPR staging).
// VDST carries the 32-bit LDS byte address (generic LDS ptr truncates to it).
static __device__ __forceinline__ void async_cp16(unsigned lds_off, const bf16* g) {
#if USE_ASYNC
  asm volatile("global_load_async_to_lds_b128 %0, %1, off"
               :: "v"(lds_off), "v"(g) : "memory");
#endif
}
static __device__ __forceinline__ void async_wait0() {
#if USE_ASYNC
  asm volatile("s_wait_asynccnt 0x0" ::: "memory");
#endif
}

// ---------------------------------------------------------------------------
// Generic bf16 WMMA GEMM. Block tile 128(M) x 64(N), BK=32, 256 threads
// (8 waves; each wave: 16 rows x 64 cols = 4 wmma accum tiles).
// Non-transposed path: double-buffered LDS, A via async global->LDS,
// B register-staged, both prefetched 2 tiles ahead.
// EPI: 0 = (+bias) write  1 = (+bias) += C (fp32 residual, in-place)
//      2 = (+bias) gelu   3 = * rowscale[row]
// ---------------------------------------------------------------------------
template<int EPI, bool OUT_BF16, bool TRANS_A>
__global__ __launch_bounds__(256) void k_gemm(
    const bf16* __restrict__ A, const bf16* __restrict__ B,
    float* __restrict__ Cf, bf16* __restrict__ Cb,
    const float* __restrict__ bias, const float* __restrict__ rowscale,
    int M, int N, int K, int lda, int ldb, int ldc,
    long long sAzo, long long sAzi, long long sBzo, long long sBzi,
    long long sCzo, long long sCzi, long long sRz, int zdiv)
{
  __shared__ __align__(16) bf16 As[2][128][40];  // stride 40 halfs = 80B (5*16)
  __shared__ __align__(16) bf16 Bs[2][64][40];   // stored transposed: Bs[n][k]
  const int t = threadIdx.x;
  const int wave = t >> 5, lane = t & 31;
  const int z = blockIdx.z, zo = z / zdiv, zi = z % zdiv;
  A += zo * sAzo + zi * sAzi;
  B += zo * sBzo + zi * sBzi;
  const long long cofs = zo * sCzo + zi * sCzi;
  if (EPI == 3) rowscale += (long long)z * sRz;
  const int m0 = blockIdx.y * 128;
  const int n0 = blockIdx.x * 64;

  v8f acc[4] = {};
  const int mA = wave * 16 + (lane & 15);
  const int kq = (lane >> 4) * 8;
  const int nl = lane & 15;
  const int kh = (lane >> 4) * 8;

  auto compute = [&](int buf) {
    v16bf a = cat8(*(const v8bf*)&As[buf][mA][kq], *(const v8bf*)&As[buf][mA][kq + 16]);
    #pragma unroll
    for (int nt = 0; nt < 4; ++nt) {
      v16bf b = cat8(*(const v8bf*)&Bs[buf][nt * 16 + nl][kh],
                     *(const v8bf*)&Bs[buf][nt * 16 + nl][kh + 16]);
      acc[nt] = __builtin_amdgcn_wmma_f32_16x16x32_bf16(
          false, a, false, b, (short)0, acc[nt], false, false);
    }
  };

  if (TRANS_A) {
    // ---- simple synchronous loop (ctx GEMM only; A memory layout [K, M]) ----
    for (int k0 = 0; k0 < K; k0 += 32) {
      {
        int kk = t >> 3, mj = (t & 7) * 16;
        const bf16* src = A + (long long)(k0 + kk) * lda + m0 + mj;
        #pragma unroll
        for (int j = 0; j < 16; ++j) As[0][mj + j][kk] = src[j];
      }
      {
        int kk = t >> 3, nj = (t & 7) * 8;
        const bf16* src = B + (long long)(k0 + kk) * ldb + n0 + nj;
        #pragma unroll
        for (int j = 0; j < 8; ++j) Bs[0][nj + j][kk] = src[j];
      }
      __syncthreads();
      compute(0);
      __syncthreads();
    }
  } else {
    // ---- async + double-buffered pipeline ----
    const int am  = t >> 1;            // A row handled by this thread
    const int asb = (t & 1) * 16;      // half-offset within the 32-half row slab
    const bf16* agp = A + (long long)(m0 + am) * lda + asb;
    const int bkk = t >> 3, bnj = (t & 7) * 8;
    const bf16* bgp = B + (long long)bkk * ldb + n0 + bnj;
    const int nk = K >> 5;

    auto issueA = [&](int kt, int buf) {
      const bf16* g = agp + kt * 32;
      unsigned l = (unsigned)(unsigned long long)&As[buf][am][asb];
      async_cp16(l, g);
      async_cp16(l + 16u, g + 8);
    };
    auto loadB = [&](int kt) -> v8bf {
      return *(const v8bf*)(bgp + (long long)kt * 32 * ldb);
    };
    auto storeB = [&](v8bf r, int buf) {
      #pragma unroll
      for (int j = 0; j < 8; ++j) Bs[buf][bnj + j][bkk] = r[j];
    };

    issueA(0, 0);
    v8bf breg = loadB(0);
    async_wait0();
    storeB(breg, 0);
    __syncthreads();

    for (int kt = 0; kt < nk; ++kt) {
      const int cur = kt & 1, nxt = cur ^ 1;
      const bool more = (kt + 1 < nk);
      if (more) {
        issueA(kt + 1, nxt);
        breg = loadB(kt + 1);
      }
      if (kt + 2 < nk) {  // global_prefetch_b8, 2 tiles ahead
        __builtin_prefetch(agp + (kt + 2) * 32, 0, 1);
        __builtin_prefetch(bgp + (long long)(kt + 2) * 32 * ldb, 0, 1);
      }
      compute(cur);
      if (more) {
        async_wait0();        // own async A(k+1) chunks landed
        __syncthreads();      // everyone done reading buf 'cur', all asyncs visible
        storeB(breg, nxt);
        __syncthreads();      // tile 'nxt' fully built
      }
    }
  }

  const int rbase = wave * 16;
  #pragma unroll
  for (int nt = 0; nt < 4; ++nt) {
    #pragma unroll
    for (int v = 0; v < 8; ++v) {
      int row = m0 + rbase + v + ((lane >> 4) << 3);  // z-local row
      int col = n0 + nt * 16 + (lane & 15);
      float val = acc[nt][v];
      if (EPI != 3) { if (bias) val += bias[col]; }
      if (EPI == 2) val = 0.5f * val * (1.0f + erff(val * 0.70710678118654752f));
      if (EPI == 3) val *= rowscale[row];
      long long ci = cofs + (long long)row * ldc + col;
      if (EPI == 1) val += Cf[ci];
      if (OUT_BF16) Cb[ci] = (bf16)val; else Cf[ci] = val;
    }
  }
}

// ---------------------------------------------------------------------------
// LayerNorm: one wave per 512-wide row, fp32 in, bf16 out
// ---------------------------------------------------------------------------
__global__ __launch_bounds__(256) void k_layernorm(
    const float* __restrict__ x, const float* __restrict__ g,
    const float* __restrict__ b, bf16* __restrict__ out)
{
  const int wave = threadIdx.x >> 5, lane = threadIdx.x & 31;
  const long long row = (long long)blockIdx.x * 8 + wave;
  const float* xr = x + row * N_DIM;
  float v[16], s = 0.f, sq = 0.f;
  #pragma unroll
  for (int i = 0; i < 16; ++i) { v[i] = xr[lane + 32 * i]; s += v[i]; sq += v[i] * v[i]; }
  #pragma unroll
  for (int off = 16; off; off >>= 1) { s += __shfl_xor(s, off, 32); sq += __shfl_xor(sq, off, 32); }
  const float mu = s * (1.f / N_DIM);
  const float rs = rsqrtf(sq * (1.f / N_DIM) - mu * mu + LNEPS);
  bf16* orow = out + row * N_DIM;
  #pragma unroll
  for (int i = 0; i < 16; ++i) {
    int c = lane + 32 * i;
    orow[c] = (bf16)((v[i] - mu) * rs * g[c] + b[c]);
  }
}

// ---------------------------------------------------------------------------
// FAVOR+ feature machinery. Block = 128 threads (4 waves), 64 seq rows.
// dd[n,m] = (q_or_k[n,:] . proj[m,:]) * dn  computed via WMMA, K = 64.
// ---------------------------------------------------------------------------
__device__ __forceinline__ void feat_load(
    const bf16* __restrict__ qkv, const bf16* __restrict__ pj,
    int n0, int b, int h, int colOff, int t,
    bf16 (*As)[80], bf16 (*Bt)[80], float* diag, bool wantDiag)
{
  {
    int m = t >> 1, kc = (t & 1) * 32;
    const bf16* p = qkv + (long long)(b * N_SEQ + n0 + m) * (3 * N_DIM) + colOff + kc;
    #pragma unroll
    for (int j = 0; j < 32; ++j) As[m][kc + j] = p[j];
  }
  for (int e = t; e < N_DH * N_MF; e += 128) {  // projT is [dh][mf]; store Bt[m][k]
    int kk = e >> 8, mm = e & 255;
    Bt[mm][kk] = pj[e];
  }
  __syncthreads();
  if (wantDiag) {
    if (t < 64) {
      float s = 0.f;
      #pragma unroll
      for (int k = 0; k < N_DH; ++k) { float q = (float)As[t][k]; s += q * q; }
      diag[t] = s * (0.5f * DNRM * DNRM);
    }
    __syncthreads();
  }
}

__device__ __forceinline__ void feat_frags(const bf16 (*As)[80], int wave, int lane,
                                           v16bf af[2])
{
  const int mA = wave * 16 + (lane & 15);
  const int kq = (lane >> 4) * 8;
  #pragma unroll
  for (int s = 0; s < 2; ++s) {
    int ks = s * 32;
    af[s] = cat8(*(const v8bf*)&As[mA][ks + kq], *(const v8bf*)&As[mA][ks + kq + 16]);
  }
}

__device__ __forceinline__ v8f dd_tile(const bf16 (*Bt)[80], const v16bf af[2],
                                       int nt, int lane)
{
  v8f acc = {};
  const int nl = nt * 16 + (lane & 15);
  const int kh = (lane >> 4) * 8;
  #pragma unroll
  for (int s = 0; s < 2; ++s) {
    int ks = s * 32;
    v16bf bb = cat8(*(const v8bf*)&Bt[nl][ks + kh], *(const v8bf*)&Bt[nl][ks + kh + 16]);
    acc = __builtin_amdgcn_wmma_f32_16x16x32_bf16(
        false, af[s], false, bb, (short)0, acc, false, false);
  }
  return acc;
}

// Pass 1 for K features: per-block max of dd -> blockmax[z*nblk + blk]
__global__ __launch_bounds__(128) void k_kmax(
    const bf16* __restrict__ qkv, const bf16* __restrict__ pj,
    float* __restrict__ blockmax)
{
  __shared__ __align__(16) bf16 As[64][80];
  __shared__ __align__(16) bf16 Bt[256][80];
  __shared__ float red[128];
  const int t = threadIdx.x, wave = t >> 5, lane = t & 31;
  const int z = blockIdx.y, b = z >> 3, h = z & 7;
  const int n0 = blockIdx.x * 64;
  feat_load(qkv, pj, n0, b, h, N_DIM + h * N_DH, t, As, Bt, nullptr, false);
  v16bf af[2]; feat_frags(As, wave, lane, af);
  float mx = -3.0e38f;
  #pragma unroll
  for (int nt = 0; nt < 16; ++nt) {
    v8f a = dd_tile(Bt, af, nt, lane);
    #pragma unroll
    for (int v = 0; v < 8; ++v) mx = fmaxf(mx, a[v]);
  }
  red[t] = mx; __syncthreads();
  for (int s = 64; s; s >>= 1) { if (t < s) red[t] = fmaxf(red[t], red[t + s]); __syncthreads(); }
  if (t == 0) blockmax[z * gridDim.x + blockIdx.x] = red[0];
}

__global__ __launch_bounds__(64) void k_redmax(
    const float* __restrict__ blockmax, float* __restrict__ kmax, int nblk)
{
  __shared__ float red[64];
  const int t = threadIdx.x;
  red[t] = blockmax[blockIdx.x * nblk + t];
  __syncthreads();
  for (int s = 32; s; s >>= 1) { if (t < s) red[t] = fmaxf(red[t], red[t + s]); __syncthreads(); }
  if (t == 0) kmax[blockIdx.x] = red[0];
}

// kp = ratio * (exp(dd - diag - kmax) + eps), streamed, bf16 out
__global__ __launch_bounds__(128) void k_kfeat(
    const bf16* __restrict__ qkv, const bf16* __restrict__ pj,
    const float* __restrict__ kmax, bf16* __restrict__ kp)
{
  __shared__ __align__(16) bf16 As[64][80];
  __shared__ __align__(16) bf16 Bt[256][80];
  __shared__ float diag[64];
  const int t = threadIdx.x, wave = t >> 5, lane = t & 31;
  const int z = blockIdx.y, b = z >> 3, h = z & 7;
  const int n0 = blockIdx.x * 64;
  feat_load(qkv, pj, n0, b, h, N_DIM + h * N_DH, t, As, Bt, diag, true);
  v16bf af[2]; feat_frags(As, wave, lane, af);
  const float km = kmax[z];
  #pragma unroll
  for (int nt = 0; nt < 16; ++nt) {
    v8f a = dd_tile(Bt, af, nt, lane);
    #pragma unroll
    for (int v = 0; v < 8; ++v) {
      int rl = wave * 16 + v + ((lane >> 4) << 3);
      float val = RATIO * (expf(a[v] - diag[rl] - km) + EPSK);
      kp[((long long)z * N_SEQ + n0 + rl) * N_MF + nt * 16 + (lane & 15)] = (bf16)val;
    }
  }
}

// qp (bf16) + d_inv = 1/(qp . k_sum): holds all 16 feature tiles in registers
__global__ __launch_bounds__(128) void k_qfeat(
    const bf16* __restrict__ qkv, const bf16* __restrict__ pj,
    const float* __restrict__ ksum, bf16* __restrict__ qp,
    float* __restrict__ dinv)
{
  __shared__ __align__(16) bf16 As[64][80];
  __shared__ __align__(16) bf16 Bt[256][80];
  __shared__ float diag[64];
  const int t = threadIdx.x, wave = t >> 5, lane = t & 31;
  const int z = blockIdx.y, b = z >> 3, h = z & 7;
  const int n0 = blockIdx.x * 64;
  feat_load(qkv, pj, n0, b, h, h * N_DH, t, As, Bt, diag, true);
  v16bf af[2]; feat_frags(As, wave, lane, af);
  v8f acc[16];
  #pragma unroll
  for (int nt = 0; nt < 16; ++nt) acc[nt] = dd_tile(Bt, af, nt, lane);
  float mx[8], dg[8], rd[8];
  #pragma unroll
  for (int v = 0; v < 8; ++v) {
    float m = acc[0][v];
    #pragma unroll
    for (int nt = 1; nt < 16; ++nt) m = fmaxf(m, acc[nt][v]);
    #pragma unroll
    for (int off = 1; off < 16; off <<= 1) m = fmaxf(m, __shfl_xor(m, off, 32));
    mx[v] = m;
    dg[v] = diag[wave * 16 + v + ((lane >> 4) << 3)];
    rd[v] = 0.f;
  }
  #pragma unroll
  for (int nt = 0; nt < 16; ++nt) {
    const float ks = ksum[z * N_MF + nt * 16 + (lane & 15)];
    #pragma unroll
    for (int v = 0; v < 8; ++v) {
      int rl = wave * 16 + v + ((lane >> 4) << 3);
      float qv = RATIO * (expf(acc[nt][v] - dg[v] - mx[v]) + EPSK);
      qp[((long long)z * N_SEQ + n0 + rl) * N_MF + nt * 16 + (lane & 15)] = (bf16)qv;
      rd[v] += qv * ks;
    }
  }
  #pragma unroll
  for (int v = 0; v < 8; ++v) {
    #pragma unroll
    for (int off = 1; off < 16; off <<= 1) rd[v] += __shfl_xor(rd[v], off, 32);
    if ((lane & 15) == 0)
      dinv[(long long)z * N_SEQ + n0 + wave * 16 + v + ((lane >> 4) << 3)] = 1.f / rd[v];
  }
}

// Two-stage column sum: stage 1 -> part[z][chunk][m], stage 2 -> ksum[z][m]
__global__ __launch_bounds__(256) void k_colsum1(
    const bf16* __restrict__ kp, float* __restrict__ part)
{
  const int z = blockIdx.x, c = blockIdx.y, m = threadIdx.x;
  const int nchunk = N_SEQ / 8;  // 512 rows per chunk
  const bf16* p = kp + ((long long)z * N_SEQ + (long long)c * nchunk) * N_MF + m;
  float s = 0.f;
  for (int n = 0; n < nchunk; ++n) s += (float)p[(long long)n * N_MF];
  part[((long long)z * 8 + c) * N_MF + m] = s;
}

__global__ __launch_bounds__(256) void k_colsum2(
    const float* __restrict__ part, float* __restrict__ ksum)
{
  const int z = blockIdx.x, m = threadIdx.x;
  float s = 0.f;
  #pragma unroll
  for (int c = 0; c < 8; ++c) s += part[((long long)z * 8 + c) * N_MF + m];
  ksum[z * N_MF + m] = s;
}

// ---------------------------------------------------------------------------
// fp32 -> bf16 conversions
// ---------------------------------------------------------------------------
__global__ __launch_bounds__(256) void k_cvt(
    const float* __restrict__ in, bf16* __restrict__ out, long long n)
{
  long long i = (long long)blockIdx.x * 256 + threadIdx.x;
  if (i < n) out[i] = (bf16)in[i];
}

// projT[l][d][m] = proj[l][m][d] * dn  (folds the dh^-0.25 normalizer)
__global__ __launch_bounds__(256) void k_projT(
    const float* __restrict__ proj, bf16* __restrict__ pt)
{
  int e = blockIdx.x * 256 + threadIdx.x;   // DEPTH*64*256 = 65536 total
  int l = e >> 14, r = e & 16383, d = r >> 8, mm = r & 255;
  pt[e] = (bf16)(proj[(l << 14) + mm * N_DH + d] * DNRM);
}

// ---------------------------------------------------------------------------
extern "C" void kernel_launch(void* const* d_in, const int* in_sizes, int n_in,
                              void* d_out, int out_size, void* d_ws, size_t ws_size,
                              hipStream_t stream) {
  const float* x    = (const float*)d_in[0];
  const float* proj = (const float*)d_in[1];
  const float* ln1g = (const float*)d_in[2];
  const float* ln1b = (const float*)d_in[3];
  const float* Wqkv = (const float*)d_in[4];
  const float* bqkv = (const float*)d_in[5];
  const float* Wo   = (const float*)d_in[6];
  const float* bo   = (const float*)d_in[7];
  const float* ln2g = (const float*)d_in[8];
  const float* ln2b = (const float*)d_in[9];
  const float* Wff1 = (const float*)d_in[10];
  const float* bff1 = (const float*)d_in[11];
  const float* Wff2 = (const float*)d_in[12];
  const float* bff2 = (const float*)d_in[13];
  float* h = (float*)d_out;   // residual stream lives in d_out (fp32)

  char* ws = (char*)d_ws;
  size_t off = 0;
  auto take = [&](size_t bytes) -> char* {
    char* p = ws + off;
    off += (bytes + 255) & ~(size_t)255;
    return p;
  };
  bf16* wqkv_bf = (bf16*)take((size_t)N_DEPTH * N_DIM * 3 * N_DIM * 2);
  bf16* wo_bf   = (bf16*)take((size_t)N_DEPTH * N_DIM * N_DIM * 2);
  bf16* wff1_bf = (bf16*)take((size_t)N_DEPTH * N_DIM * N_FF * 2);
  bf16* wff2_bf = (bf16*)take((size_t)N_DEPTH * N_FF * N_DIM * 2);
  bf16* pt_bf   = (bf16*)take((size_t)N_DEPTH * N_DH * N_MF * 2);
  bf16* lnbuf   = (bf16*)take((size_t)N_ROWS * N_DIM * 2);
  bf16* qkvbuf  = (bf16*)take((size_t)N_ROWS * 3 * N_DIM * 2);
  bf16* kqbuf   = (bf16*)take((size_t)16 * N_SEQ * N_MF * 2);  // kp, then qp, then ff
  bf16* ctxbuf  = (bf16*)take((size_t)16 * N_MF * N_DH * 2);
  float* dinv   = (float*)take((size_t)16 * N_SEQ * 4);
  float* ksum   = (float*)take((size_t)16 * N_MF * 4);
  float* kmax   = (float*)take((size_t)16 * 4);
  float* bmax   = (float*)take((size_t)16 * 64 * 4);
  float* cpart  = (float*)take((size_t)16 * 8 * N_MF * 4);
  bf16* attnbuf = (bf16*)take((size_t)N_ROWS * N_DIM * 2);

  hipMemcpyAsync(h, x, (size_t)N_ROWS * N_DIM * 4, hipMemcpyDeviceToDevice, stream);

  auto cvt = [&](const float* src, bf16* dst, long long n) {
    k_cvt<<<dim3((unsigned)((n + 255) / 256)), 256, 0, stream>>>(src, dst, n);
  };
  cvt(Wqkv, wqkv_bf, (long long)N_DEPTH * N_DIM * 3 * N_DIM);
  cvt(Wo,   wo_bf,   (long long)N_DEPTH * N_DIM * N_DIM);
  cvt(Wff1, wff1_bf, (long long)N_DEPTH * N_DIM * N_FF);
  cvt(Wff2, wff2_bf, (long long)N_DEPTH * N_FF * N_DIM);
  k_projT<<<256, 256, 0, stream>>>(proj, pt_bf);

  for (int i = 0; i < N_DEPTH; ++i) {
    const bf16* wq = wqkv_bf + (long long)i * N_DIM * 3 * N_DIM;
    const bf16* pt = pt_bf + (long long)i * N_DH * N_MF;

    // ---- LN1 + QKV (bf16 out) ----
    k_layernorm<<<N_ROWS / 8, 256, 0, stream>>>(h, ln1g + i * N_DIM, ln1b + i * N_DIM, lnbuf);
    k_gemm<0, true, false><<<dim3(3 * N_DIM / 64, N_ROWS / 128, 1), 256, 0, stream>>>(
        lnbuf, wq, nullptr, qkvbuf, bqkv + i * 3 * N_DIM, nullptr,
        N_ROWS, 3 * N_DIM, N_DIM, N_DIM, 3 * N_DIM, 3 * N_DIM,
        0, 0, 0, 0, 0, 0, 0, 1);

    // ---- FAVOR+ K features ----
    k_kmax<<<dim3(N_SEQ / 64, 16), 128, 0, stream>>>(qkvbuf, pt, bmax);
    k_redmax<<<16, 64, 0, stream>>>(bmax, kmax, N_SEQ / 64);
    k_kfeat<<<dim3(N_SEQ / 64, 16), 128, 0, stream>>>(qkvbuf, pt, kmax, kqbuf);
    k_colsum1<<<dim3(16, 8), 256, 0, stream>>>(kqbuf, cpart);
    k_colsum2<<<16, 256, 0, stream>>>(cpart, ksum);

    // ---- ctx[z] = kp^T @ v : M=256, N=64, K=4096, z=(b,h) ----
    k_gemm<0, true, true><<<dim3(1, N_MF / 128, 16), 256, 0, stream>>>(
        kqbuf, qkvbuf + 2 * N_DIM, nullptr, ctxbuf, nullptr, nullptr,
        N_MF, N_DH, N_SEQ, N_MF, 3 * N_DIM, N_DH,
        (long long)8 * N_SEQ * N_MF, (long long)N_SEQ * N_MF,
        (long long)N_SEQ * 3 * N_DIM, (long long)N_DH,
        (long long)8 * N_MF * N_DH, (long long)N_MF * N_DH, 0, 8);

    // ---- FAVOR+ Q features + d_inv (qp overwrites kp; ctx already consumed it) ----
    k_qfeat<<<dim3(N_SEQ / 64, 16), 128, 0, stream>>>(qkvbuf, pt, ksum, kqbuf, dinv);

    // ---- attn[z] = (qp @ ctx) * d_inv -> merged-head bf16 [8192,512] ----
    k_gemm<3, true, false><<<dim3(1, N_SEQ / 128, 16), 256, 0, stream>>>(
        kqbuf, ctxbuf, nullptr, attnbuf, nullptr, dinv,
        N_SEQ, N_DH, N_MF, N_MF, N_DH, N_DIM,
        (long long)8 * N_SEQ * N_MF, (long long)N_SEQ * N_MF,
        (long long)8 * N_MF * N_DH, (long long)N_MF * N_DH,
        (long long)N_SEQ * N_DIM, (long long)N_DH, (long long)N_SEQ, 8);

    // ---- h += attn @ Wo + bo (fp32 residual in-place) ----
    k_gemm<1, false, false><<<dim3(N_DIM / 64, N_ROWS / 128, 1), 256, 0, stream>>>(
        attnbuf, wo_bf + (long long)i * N_DIM * N_DIM, h, nullptr, bo + i * N_DIM, nullptr,
        N_ROWS, N_DIM, N_DIM, N_DIM, N_DIM, N_DIM, 0, 0, 0, 0, 0, 0, 0, 1);

    // ---- LN2 + FFN ----
    k_layernorm<<<N_ROWS / 8, 256, 0, stream>>>(h, ln2g + i * N_DIM, ln2b + i * N_DIM, lnbuf);
    k_gemm<2, true, false><<<dim3(N_FF / 64, N_ROWS / 128, 1), 256, 0, stream>>>(
        lnbuf, wff1_bf + (long long)i * N_DIM * N_FF, nullptr, kqbuf, bff1 + i * N_FF, nullptr,
        N_ROWS, N_FF, N_DIM, N_DIM, N_FF, N_FF, 0, 0, 0, 0, 0, 0, 0, 1);
    k_gemm<1, false, false><<<dim3(N_DIM / 64, N_ROWS / 128, 1), 256, 0, stream>>>(
        kqbuf, wff2_bf + (long long)i * N_FF * N_DIM, h, nullptr, bff2 + i * N_DIM, nullptr,
        N_ROWS, N_DIM, N_FF, N_FF, N_DIM, N_DIM, 0, 0, 0, 0, 0, 0, 0, 1);
  }
}
